// TextMoE_56118042689567
// MI455X (gfx1250) — matrix operand order repaired and checked by reading, not compiled, not run on hardware
//
#include <hip/hip_runtime.h>
#include <hip/hip_bf16.h>

// ---------------------------------------------------------------------------
// TextMoE forward for MI455X (gfx1250, wave32, WMMA).
//   B=16 S=1024 D=1024 H=8 HS=128 E=10 K=2 DFF=4096 NCLS=10
// All GEMMs run on v_wmma_f32_16x16x32_f16 (f16 in, f32 accumulate).
// fp32 weights are converted to f16 in-register during LDS staging (one HBM
// read per weight element). MoE is a gate-masked dense sweep over experts
// (mathematically exact since non-top-2 gates are 0) with per-tile skip.
// GEMM uses double-buffered LDS (one barrier per K-step) and
// GLOBAL_LOAD_ASYNC_TO_LDS_B128 for the f16 A tile, retired with
// s_wait_asynccnt before the publishing barrier.
// ---------------------------------------------------------------------------

typedef __attribute__((ext_vector_type(16))) _Float16 v16h;
typedef __attribute__((ext_vector_type(8)))  _Float16 v8h;
typedef __attribute__((ext_vector_type(8)))  float    v8f;

#define NB    16
#define SS    1024
#define DD    1024
#define HH    8
#define HSZ   128
#define EE    10
#define DFF_  4096
#define NCLS_ 10
#define NTOK  (NB * SS)

enum { MODE_QKV = 0, MODE_OPROJ = 1, MODE_FFN1 = 2, MODE_FFN2 = 3 };

// GEMM block tiling: 64(M) x 128(N), K chunk 32, 8 waves of 2x2 16x16 tiles.
#define BM  64
#define BN  128
#define BKK 32
#define APITCH 40   // BK + 8 halfs of padding (16B aligned rows)
#define BPITCH 40

// ---- CDNA5 async global->LDS path (guarded; falls back to load+store) -----
#if defined(__has_builtin)
#  if __has_builtin(__builtin_amdgcn_global_load_async_to_lds_b128)
#    define HAVE_ASYNC_LDS 1
#  endif
#endif
#ifndef HAVE_ASYNC_LDS
#  define HAVE_ASYNC_LDS 0
#endif

// The builtin takes pointers to GCC-vector int4 in AS(1) (global) / AS(3) (LDS).
typedef int i32x4 __attribute__((vector_size(16)));
typedef __attribute__((address_space(1))) i32x4 gas_i32x4;
typedef __attribute__((address_space(3))) i32x4 las_i32x4;

__device__ __forceinline__ void async_copy_b128(const _Float16* g, _Float16* l) {
#if HAVE_ASYNC_LDS
  __builtin_amdgcn_global_load_async_to_lds_b128((gas_i32x4*)g, (las_i32x4*)l,
                                                 0, 0);
#else
  *(v8h*)l = *(const v8h*)g;
#endif
}

__device__ __forceinline__ void async_wait_all() {
#if HAVE_ASYNC_LDS
#  if __has_builtin(__builtin_amdgcn_s_wait_asynccnt)
  __builtin_amdgcn_s_wait_asynccnt(0);
#  else
  asm volatile("s_wait_asynccnt 0" ::: "memory");
#  endif
#endif
}

__device__ __forceinline__ v8f v8f_zero() {
  v8f z;
#pragma unroll
  for (int i = 0; i < 8; ++i) z[i] = 0.f;
  return z;
}

// Build a v16h fragment from two contiguous 8-half (16B) groups.
__device__ __forceinline__ v16h frag_pair(const _Float16* lo, const _Float16* hi) {
  v8h a = *(const v8h*)lo;
  v8h b = *(const v8h*)hi;
  return __builtin_shufflevector(a, b, 0, 1, 2, 3, 4, 5, 6, 7,
                                       8, 9, 10, 11, 12, 13, 14, 15);
}

__device__ __forceinline__ v8f wmma_f16(v16h a, v16h b, v8f c) {
  return __builtin_amdgcn_wmma_f32_16x16x32_f16(false, a, false, b,
                                                (short)0, c, false, false);
}

// ---------------------------------------------------------------------------
// Embedding gather: x[row, :] = emb[ids[row], :]
// ---------------------------------------------------------------------------
__global__ __launch_bounds__(256)
void k_embed(const int* __restrict__ ids, const float* __restrict__ emb,
             float* __restrict__ x) {
  const int row = blockIdx.x;
  const int tid = threadIdx.x;
  const int id = ids[row];
  float4 v = *(const float4*)(emb + (size_t)id * DD + tid * 4);
  *(float4*)(x + (size_t)row * DD + tid * 4) = v;
}

// ---------------------------------------------------------------------------
// LayerNorm over D=1024, one block per row, writes f16 for the WMMA path.
// ---------------------------------------------------------------------------
__global__ __launch_bounds__(256)
void k_ln(const float* __restrict__ x, const float* __restrict__ s,
          const float* __restrict__ b, _Float16* __restrict__ out) {
  __shared__ float rs[256], rq[256];
  const int row = blockIdx.x, tid = threadIdx.x;
  const float* xr = x + (size_t)row * DD;
  float4 xv = *(const float4*)(xr + tid * 4);
  float sum = xv.x + xv.y + xv.z + xv.w;
  float sq  = xv.x * xv.x + xv.y * xv.y + xv.z * xv.z + xv.w * xv.w;
  rs[tid] = sum; rq[tid] = sq;
  __syncthreads();
  for (int off = 128; off > 0; off >>= 1) {
    if (tid < off) { rs[tid] += rs[tid + off]; rq[tid] += rq[tid + off]; }
    __syncthreads();
  }
  const float mean = rs[0] * (1.f / (float)DD);
  const float var  = rq[0] * (1.f / (float)DD) - mean * mean;
  const float rstd = rsqrtf(var + 1e-5f);
  const int i = tid * 4;
  _Float16* o = out + (size_t)row * DD + i;
  o[0] = (_Float16)((xv.x - mean) * rstd * s[i + 0] + b[i + 0]);
  o[1] = (_Float16)((xv.y - mean) * rstd * s[i + 1] + b[i + 1]);
  o[2] = (_Float16)((xv.z - mean) * rstd * s[i + 2] + b[i + 2]);
  o[3] = (_Float16)((xv.w - mean) * rstd * s[i + 3] + b[i + 3]);
}

// ---------------------------------------------------------------------------
// Generic WMMA GEMM: C[M,N] = A[M,K](f16) * B[K,N](f32->f16), fused epilogues.
// Double-buffered LDS, async A staging, packed-b32 transposed B staging.
// ---------------------------------------------------------------------------
__global__ __launch_bounds__(256)
void k_gemm_f16(const _Float16* __restrict__ A, const float* __restrict__ Bw,
                int M, int N, int K, int mode,
                const float* __restrict__ bias,
                float* __restrict__ outf, _Float16* __restrict__ outh,
                _Float16* __restrict__ q16, _Float16* __restrict__ k16,
                _Float16* __restrict__ v16,
                const float* __restrict__ pos,
                const float* __restrict__ gate, int gstride, int beta) {
  __shared__ _Float16 As[2][BM * APITCH];
  __shared__ _Float16 Bt[2][BN * BPITCH];   // transposed: [n][k]

  const int tid = threadIdx.x;
  const int m0 = blockIdx.y * BM;
  const int n0 = blockIdx.x * BN;

  // Expert-tile skip: if no token in this 64-row tile routed to this expert,
  // the whole tile contributes exactly zero.
  if (mode == MODE_FFN1 || mode == MODE_FFN2) {
    float gmax = 0.f;
    for (int i = 0; i < BM; ++i)
      gmax = fmaxf(gmax, gate[(size_t)(m0 + i) * gstride]);
    if (gmax <= 0.f) {
      if (mode == MODE_FFN2 && beta == 0) {
        for (int i = tid; i < BM * (BN / 4); i += 256) {
          int r = i >> 5, c = (i & 31) << 2;
          *(float4*)&outf[(size_t)(m0 + r) * N + n0 + c] =
              make_float4(0.f, 0.f, 0.f, 0.f);
        }
      }
      return;
    }
  }

  const int wave = tid >> 5, lane = tid & 31;
  const int wm = wave >> 2, wn = wave & 3;      // 2 x 4 wave grid
  const int ls = lane & 15, hsel = lane >> 4;

  v8f acc[2][2];
#pragma unroll
  for (int i = 0; i < 2; ++i)
#pragma unroll
    for (int j = 0; j < 2; ++j) acc[i][j] = v8f_zero();

  const int ar = tid >> 2, ac = (tid & 3) << 3;     // A: 16B per thread
  const int k2 = (tid >> 4) << 1, nB = (tid & 15) << 3;  // B: 2 K-rows x 8 N
  const _Float16* Ag = A + (size_t)(m0 + ar) * K + ac;
  const float*    Bg = Bw + (size_t)k2 * N + n0 + nB;

  // Stage one K-chunk into LDS buffer `buf`:
  //  - A tile: async DMA (ASYNCcnt) of 16B f16 per thread
  //  - B tile: fp32 loads, cvt to f16, (k,k+1) pairs packed into b32 stores
  auto stage = [&](int buf, int kk) {
    async_copy_b128(Ag + kk, &As[buf][ar * APITCH + ac]);
    const float* bp = Bg + (size_t)kk * N;
    float4 a0 = *(const float4*)(bp);
    float4 a1 = *(const float4*)(bp + 4);
    float4 b0 = *(const float4*)(bp + N);
    float4 b1 = *(const float4*)(bp + N + 4);
    const float la[8] = {a0.x, a0.y, a0.z, a0.w, a1.x, a1.y, a1.z, a1.w};
    const float lb[8] = {b0.x, b0.y, b0.z, b0.w, b1.x, b1.y, b1.z, b1.w};
#pragma unroll
    for (int j = 0; j < 8; ++j) {
      union { _Float16 h[2]; unsigned int u; } p;
      p.h[0] = (_Float16)la[j];
      p.h[1] = (_Float16)lb[j];
      *(unsigned int*)&Bt[buf][(nB + j) * BPITCH + k2] = p.u;
    }
    if (kk + BKK < K) {   // CDNA5 global prefetch of the chunk after next
      __builtin_prefetch(Ag + kk + BKK, 0, 1);
      __builtin_prefetch(bp + (size_t)BKK * N, 0, 1);
    }
  };

  stage(0, 0);
  int cur = 0;
  for (int kk = 0; kk < K; kk += BKK) {
    async_wait_all();      // this wave's async DMA into `cur` has landed
    __syncthreads();       // all waves published `cur`
    if (kk + BKK < K) stage(cur ^ 1, kk + BKK);  // overlap with WMMA below

    const _Float16* Asb = &As[cur][0];
    const _Float16* Btb = &Bt[cur][0];
    v16h bf[2];
#pragma unroll
    for (int j = 0; j < 2; ++j) {
      // B frag: half-wave K interleave {0-7,16-23} / {8-15,24-31}.
      const _Float16* p = &Btb[(wn * 32 + j * 16 + ls) * BPITCH + (hsel << 3)];
      bf[j] = frag_pair(p, p + 16);
    }
#pragma unroll
    for (int i = 0; i < 2; ++i) {
      // A frag: half-wave K interleave {0-7,16-23} / {8-15,24-31}.
      const _Float16* p = &Asb[(wm * 32 + i * 16 + ls) * APITCH + (hsel << 3)];
      v16h af = frag_pair(p, p + 16);
      acc[i][0] = wmma_f16(af, bf[0], acc[i][0]);
      acc[i][1] = wmma_f16(af, bf[1], acc[i][1]);
    }
    cur ^= 1;
  }

  // Epilogue. C layout: lane<16 elem i -> (M=i, N=lane); lane>=16 -> (M=8+i).
#pragma unroll
  for (int i = 0; i < 2; ++i)
#pragma unroll
    for (int j = 0; j < 2; ++j) {
      const int rb = m0 + wm * 32 + i * 16 + hsel * 8;
      const int cb = n0 + wn * 32 + j * 16 + ls;
#pragma unroll
      for (int r = 0; r < 8; ++r) {
        const float v = acc[i][j][r];
        const int row = rb + r, col = cb;
        if (mode == MODE_QKV) {
          const int which = col >> 10, d0 = col & 1023;
          const int h = d0 >> 7, hs = d0 & 127;
          const int bb = row >> 10, sI = row & 1023;
          _Float16* dst = (which == 0) ? q16 : ((which == 1) ? k16 : v16);
          dst[(((size_t)bb * HH + h) * SS + sI) * HSZ + hs] = (_Float16)v;
        } else if (mode == MODE_OPROJ) {
          const size_t o = (size_t)row * N + col;
          outf[o] = v + bias[col] + outf[o] +
                    pos[(size_t)(row & (SS - 1)) * N + col];
        } else if (mode == MODE_FFN1) {
          outh[(size_t)row * N + col] = (_Float16)fmaxf(v + bias[col], 0.f);
        } else {  // MODE_FFN2
          const float g = gate[(size_t)row * gstride];
          const float rr = g * (v + bias[col]);
          const size_t o = (size_t)row * N + col;
          outf[o] = beta ? (outf[o] + rr) : rr;
        }
      }
    }
}

// ---------------------------------------------------------------------------
// Flash attention, one block per (b*H, 128-query tile). 8 waves x 16 q rows.
// ---------------------------------------------------------------------------
#define QT  128
#define KTT 32

__global__ __launch_bounds__(256)
void k_attn(const _Float16* __restrict__ q16, const _Float16* __restrict__ k16,
            const _Float16* __restrict__ v16, _Float16* __restrict__ o16) {
  __shared__ _Float16 Vt[HSZ * KTT];        // [hs][key]
  __shared__ _Float16 Ps[8 * 16 * KTT];     // per-wave P tile [16][32]

  const int tid = threadIdx.x;
  const int wave = tid >> 5, lane = tid & 31;
  const int ls = lane & 15, hsel = lane >> 4;
  const int bh = blockIdx.y, qt = blockIdx.x;

  const _Float16* Qb = q16 + (size_t)bh * SS * HSZ;
  const _Float16* Kb = k16 + (size_t)bh * SS * HSZ;
  const _Float16* Vb = v16 + (size_t)bh * SS * HSZ;
  const int qrow0 = qt * QT + wave * 16;

  // Preload this wave's Q A-fragments (4 K-chunks of 32) from global.
  v16h aq[4];
  {
    const _Float16* qp = Qb + (size_t)(qrow0 + ls) * HSZ + hsel * 8;
#pragma unroll
    for (int c = 0; c < 4; ++c) aq[c] = frag_pair(qp + c * 32, qp + c * 32 + 16);
  }

  v8f Of[8];
#pragma unroll
  for (int t = 0; t < 8; ++t) Of[t] = v8f_zero();
  float mrow[8], lrow[8];
#pragma unroll
  for (int i = 0; i < 8; ++i) { mrow[i] = -3.0e38f; lrow[i] = 0.f; }

  const int nkt = qt * 4 + 4;               // causal key-tile bound
  const int keyP = (tid & 15) << 1;         // 2 keys per thread
  const int hsV  = (tid >> 4) << 3;         // 8 hs per thread

  for (int kt = 0; kt < nkt; ++kt) {
    const int key0 = kt * KTT;
    __syncthreads();
    {  // cooperative transpose-stage of V tile: Vt[hs][key], b32-packed pairs
      const _Float16* vp0 = Vb + (size_t)(key0 + keyP) * HSZ + hsV;
      v8h a = *(const v8h*)vp0, b = *(const v8h*)(vp0 + HSZ);
#pragma unroll
      for (int j = 0; j < 8; ++j) {
        union { _Float16 h[2]; unsigned int u; } p;
        p.h[0] = a[j];
        p.h[1] = b[j];
        *(unsigned int*)&Vt[(hsV + j) * KTT + keyP] = p.u;
      }
    }
    __syncthreads();
    if (kt + 1 < nkt)
      __builtin_prefetch(Vb + (size_t)(key0 + KTT + keyP) * HSZ + hsV, 0, 1);
    if (key0 > qrow0 + 15) continue;        // fully-masked tile for this wave

    // S = Q K^T (two 16-key halves); K row-major == B^T fragment layout.
    v8f sc[2];
#pragma unroll
    for (int hn = 0; hn < 2; ++hn) {
      sc[hn] = v8f_zero();
      const _Float16* kp =
          Kb + (size_t)(key0 + hn * 16 + ls) * HSZ + hsel * 8;
#pragma unroll
      for (int c = 0; c < 4; ++c) {
        v16h bk = frag_pair(kp + c * 32, kp + c * 32 + 16);
        sc[hn] = wmma_f16(aq[c], bk, sc[hn]);
      }
    }
    // Scale + causal mask.
    const int qr = qrow0 + hsel * 8;
#pragma unroll
    for (int hn = 0; hn < 2; ++hn)
#pragma unroll
      for (int i = 0; i < 8; ++i) {
        const int key = key0 + hn * 16 + ls;
        const float sv = sc[hn][i] * 0.08838834764831845f;  // HS^-0.5
        sc[hn][i] = (key <= qr + i) ? sv : -3.0e38f;
      }
    // Row max (16-lane xor reductions within each half-wave).
    float mx[8], corr[8];
#pragma unroll
    for (int i = 0; i < 8; ++i) mx[i] = fmaxf(sc[0][i], sc[1][i]);
#pragma unroll
    for (int off = 1; off < 16; off <<= 1)
#pragma unroll
      for (int i = 0; i < 8; ++i)
        mx[i] = fmaxf(mx[i], __shfl_xor(mx[i], off, 32));
#pragma unroll
    for (int i = 0; i < 8; ++i) {
      const float mn = fmaxf(mrow[i], mx[i]);
      corr[i] = __expf(mrow[i] - mn);
      mrow[i] = mn;
    }
    // P = exp(S - m), row sums, rescale O.
#pragma unroll
    for (int hn = 0; hn < 2; ++hn)
#pragma unroll
      for (int i = 0; i < 8; ++i) sc[hn][i] = __expf(sc[hn][i] - mrow[i]);
#pragma unroll
    for (int i = 0; i < 8; ++i) {
      float ps = sc[0][i] + sc[1][i];
#pragma unroll
      for (int off = 1; off < 16; off <<= 1) ps += __shfl_xor(ps, off, 32);
      lrow[i] = lrow[i] * corr[i] + ps;
    }
#pragma unroll
    for (int t = 0; t < 8; ++t)
#pragma unroll
      for (int i = 0; i < 8; ++i) Of[t][i] *= corr[i];

    // C-layout -> A-layout relayout of P via per-wave LDS (in-order DS ops).
    _Float16* Pw = &Ps[wave * 16 * KTT];
#pragma unroll
    for (int hn = 0; hn < 2; ++hn)
#pragma unroll
      for (int i = 0; i < 8; ++i)
        Pw[(hsel * 8 + i) * KTT + hn * 16 + ls] = (_Float16)sc[hn][i];
    __builtin_amdgcn_wave_barrier();
    v16h ap;
    {
      const _Float16* pp = Pw + (size_t)ls * KTT + hsel * 8;
      ap = frag_pair(pp, pp + 16);
    }
    // O += P V  (8 column tiles of 16 over HS=128).
#pragma unroll
    for (int nt = 0; nt < 8; ++nt) {
      const _Float16* vtp = &Vt[(nt * 16 + ls) * KTT + hsel * 8];
      v16h bv = frag_pair(vtp, vtp + 16);
      Of[nt] = wmma_f16(ap, bv, Of[nt]);
    }
  }

  // Normalize and write [token, h*HS + hs] f16.
  const int bb = bh / HH, h = bh % HH;
#pragma unroll
  for (int i = 0; i < 8; ++i) {
    const float inv = 1.0f / lrow[i];
    const int q = qrow0 + hsel * 8 + i;
    const size_t base = ((size_t)bb * SS + q) * DD + h * HSZ;
#pragma unroll
    for (int nt = 0; nt < 8; ++nt)
      o16[base + nt * 16 + ls] = (_Float16)(Of[nt][i] * inv);
  }
}

// ---------------------------------------------------------------------------
// Router: logits = LN(x) @ wg + bg; top-2 softmax gates (exact sparse gates).
// ---------------------------------------------------------------------------
__global__ __launch_bounds__(256)
void k_gate(const _Float16* __restrict__ xb, const float* __restrict__ wg,
            const float* __restrict__ bg, float* __restrict__ gate) {
  const int t = blockIdx.x * 256 + threadIdx.x;
  if (t >= NTOK) return;
  float acc[EE];
#pragma unroll
  for (int e = 0; e < EE; ++e) acc[e] = bg[e];
  const _Float16* xr = xb + (size_t)t * DD;
  for (int d = 0; d < DD; ++d) {
    const float xv = (float)xr[d];
    const float* wr = wg + (size_t)d * EE;
#pragma unroll
    for (int e = 0; e < EE; ++e) acc[e] += xv * wr[e];
  }
  int i1 = 0;
#pragma unroll
  for (int e = 1; e < EE; ++e) if (acc[e] > acc[i1]) i1 = e;
  int i2 = (i1 == 0) ? 1 : 0;
#pragma unroll
  for (int e = 0; e < EE; ++e)
    if (e != i1 && acc[e] > acc[i2]) i2 = e;
  const float ex = __expf(acc[i2] - acc[i1]);
  const float inv = 1.f / (1.f + ex);
  float* gr = gate + (size_t)t * EE;
#pragma unroll
  for (int e = 0; e < EE; ++e) gr[e] = 0.f;
  gr[i1] = inv;
  gr[i2] = ex * inv;
}

// ---------------------------------------------------------------------------
// feat[b,d] = mean_s second[b,s,d]
// ---------------------------------------------------------------------------
__global__ __launch_bounds__(256)
void k_mean(const float* __restrict__ second, float* __restrict__ feat) {
  const int i = blockIdx.x * 256 + threadIdx.x;  // b*D + d
  const int b = i >> 10, d = i & 1023;
  float acc = 0.f;
  for (int s = 0; s < SS; ++s)
    acc += second[((size_t)b * SS + s) * DD + d];
  feat[i] = acc * (1.f / (float)SS);
}

// ---------------------------------------------------------------------------
// cls[b,c] = feat[b,:] @ wc[:,c] + bc[c]   (16 x 10, tiny fp32 kernel)
// ---------------------------------------------------------------------------
__global__ __launch_bounds__(256)
void k_cls(const float* __restrict__ feat, const float* __restrict__ wc,
           const float* __restrict__ bc, float* __restrict__ cls) {
  const int i = threadIdx.x;
  if (i >= NB * NCLS_) return;
  const int b = i / NCLS_, c = i % NCLS_;
  float acc = bc[c];
  for (int d = 0; d < DD; ++d)
    acc += feat[(size_t)b * DD + d] * wc[(size_t)d * NCLS_ + c];
  cls[i] = acc;
}

// ---------------------------------------------------------------------------
extern "C" void kernel_launch(void* const* d_in, const int* in_sizes, int n_in,
                              void* d_out, int out_size, void* d_ws,
                              size_t ws_size, hipStream_t stream) {
  (void)in_sizes; (void)n_in; (void)out_size; (void)ws_size;

  const int*   ids  = (const int*)  d_in[0];
  const float* emb  = (const float*)d_in[2];
  const float* pos  = (const float*)d_in[3];
  const float* ln1s = (const float*)d_in[4];
  const float* ln1b = (const float*)d_in[5];
  const float* ln2s = (const float*)d_in[6];
  const float* ln2b = (const float*)d_in[7];
  const float* ln3s = (const float*)d_in[8];
  const float* ln3b = (const float*)d_in[9];
  const float* wqkv = (const float*)d_in[10];
  const float* wo   = (const float*)d_in[11];
  const float* bo   = (const float*)d_in[12];
  const float* wg1  = (const float*)d_in[13];
  const float* bg1  = (const float*)d_in[14];
  const float* w1a  = (const float*)d_in[15];
  const float* b1a  = (const float*)d_in[16];
  const float* w2a  = (const float*)d_in[17];
  const float* b2a  = (const float*)d_in[18];
  const float* wg2  = (const float*)d_in[19];
  const float* bg2  = (const float*)d_in[20];
  const float* w1b  = (const float*)d_in[21];
  const float* b1b  = (const float*)d_in[22];
  const float* w2b  = (const float*)d_in[23];
  const float* b2b  = (const float*)d_in[24];
  const float* wc   = (const float*)d_in[25];
  const float* bc   = (const float*)d_in[26];

  float* outF   = (float*)d_out;
  float* first  = outF;
  float* second = outF + (size_t)NTOK * DD;
  float* feat   = second + (size_t)NTOK * DD;
  float* cls    = feat + (size_t)NB * DD;

  char* wsp = (char*)d_ws;
  size_t off = 0;
  auto alloc = [&](size_t bytes) -> char* {
    char* p = wsp + off;
    off = (off + bytes + 255) & ~(size_t)255;
    return p;
  };
  float*    x    = (float*)   alloc((size_t)NTOK * DD * sizeof(float));
  _Float16* xb   = (_Float16*)alloc((size_t)NTOK * DD * sizeof(_Float16));
  _Float16* q16  = (_Float16*)alloc((size_t)NTOK * DD * sizeof(_Float16));
  _Float16* k16  = (_Float16*)alloc((size_t)NTOK * DD * sizeof(_Float16));
  _Float16* v16  = (_Float16*)alloc((size_t)NTOK * DD * sizeof(_Float16));
  _Float16* ao16 = (_Float16*)alloc((size_t)NTOK * DD * sizeof(_Float16));
  _Float16* H16  = (_Float16*)alloc((size_t)NTOK * DFF_ * sizeof(_Float16));
  float*    g1   = (float*)   alloc((size_t)NTOK * EE * sizeof(float));
  float*    g2   = (float*)   alloc((size_t)NTOK * EE * sizeof(float));

  // 1. Embedding gather.
  k_embed<<<NTOK, 256, 0, stream>>>(ids, emb, x);

  // 2. Attention block: LN1 -> QKV -> flash attention -> out-proj
  //    (epilogue fuses +bo + residual + pos_emb into x).
  k_ln<<<NTOK, 256, 0, stream>>>(x, ln1s, ln1b, xb);
  k_gemm_f16<<<dim3(3 * DD / BN, NTOK / BM), 256, 0, stream>>>(
      xb, wqkv, NTOK, 3 * DD, DD, MODE_QKV,
      nullptr, nullptr, nullptr, q16, k16, v16, nullptr, nullptr, 0, 0);
  k_attn<<<dim3(SS / QT, NB * HH), 256, 0, stream>>>(q16, k16, v16, ao16);
  k_gemm_f16<<<dim3(DD / BN, NTOK / BM), 256, 0, stream>>>(
      ao16, wo, NTOK, DD, DD, MODE_OPROJ,
      bo, x, nullptr, nullptr, nullptr, nullptr, pos, nullptr, 0, 0);

  // 3. MoE layer 1 -> first.
  k_ln<<<NTOK, 256, 0, stream>>>(x, ln2s, ln2b, xb);
  k_gate<<<NTOK / 256, 256, 0, stream>>>(xb, wg1, bg1, g1);
  for (int e = 0; e < EE; ++e) {
    k_gemm_f16<<<dim3(DFF_ / BN, NTOK / BM), 256, 0, stream>>>(
        xb, w1a + (size_t)e * DD * DFF_, NTOK, DFF_, DD, MODE_FFN1,
        b1a + (size_t)e * DFF_, nullptr, H16, nullptr, nullptr, nullptr,
        nullptr, g1 + e, EE, 0);
    k_gemm_f16<<<dim3(DD / BN, NTOK / BM), 256, 0, stream>>>(
        H16, w2a + (size_t)e * DFF_ * DD, NTOK, DD, DFF_, MODE_FFN2,
        b2a + (size_t)e * DD, first, nullptr, nullptr, nullptr, nullptr,
        nullptr, g1 + e, EE, (e > 0) ? 1 : 0);
  }

  // 4. MoE layer 2 -> second.
  k_ln<<<NTOK, 256, 0, stream>>>(x, ln3s, ln3b, xb);
  k_gate<<<NTOK / 256, 256, 0, stream>>>(xb, wg2, bg2, g2);
  for (int e = 0; e < EE; ++e) {
    k_gemm_f16<<<dim3(DFF_ / BN, NTOK / BM), 256, 0, stream>>>(
        xb, w1b + (size_t)e * DD * DFF_, NTOK, DFF_, DD, MODE_FFN1,
        b1b + (size_t)e * DFF_, nullptr, H16, nullptr, nullptr, nullptr,
        nullptr, g2 + e, EE, 0);
    k_gemm_f16<<<dim3(DD / BN, NTOK / BM), 256, 0, stream>>>(
        H16, w2b + (size_t)e * DFF_ * DD, NTOK, DD, DFF_, MODE_FFN2,
        b2b + (size_t)e * DD, second, nullptr, nullptr, nullptr, nullptr,
        nullptr, g2 + e, EE, (e > 0) ? 1 : 0);
  }

  // 5. Pooled features + classifier head.
  k_mean<<<(NB * DD) / 256, 256, 0, stream>>>(second, feat);
  k_cls<<<1, 256, 0, stream>>>(feat, wc, bc, cls);
}
